// NumEncoderTransformer_50525995270541
// MI455X (gfx1250) — matrix operand order repaired and checked by reading, compiled, not run on hardware
//
#include <hip/hip_runtime.h>

// ---------------------------------------------------------------------------
// KAN 2-layer forward on gfx1250 (MI455X), f16 WMMA with f32 accumulation,
// TDM (tensor_load_to_lds) streaming of all weight panels, double buffering.
// B=65536, F=32, H=64, OUT=4096, C=8 spline coeffs, K1=288, K2=576.
// ---------------------------------------------------------------------------

#define B_ROWS 65536
#define F_IN   32
#define H_MID  64
#define N_OUT  4096
#define K1     288   // 32 + 32*8
#define K2     576   // 64 + 64*8

typedef _Float16 half16_t __attribute__((ext_vector_type(16)));
typedef _Float16 half8_t  __attribute__((ext_vector_type(8)));
typedef float    f32x8    __attribute__((ext_vector_type(8)));
typedef unsigned int u32x4 __attribute__((ext_vector_type(4)));
typedef int      i32x4    __attribute__((ext_vector_type(4)));
typedef int      i32x8    __attribute__((ext_vector_type(8)));

__device__ __forceinline__ half16_t cat8(half8_t lo, half8_t hi) {
  return __builtin_shufflevector(lo, hi, 0,1,2,3,4,5,6,7,8,9,10,11,12,13,14,15);
}

__device__ __forceinline__ float silu_f(float x) {
  return x / (1.0f + __expf(-x));
}

// Cubic B-spline bases on the uniform KAN grid g[j] = 0.4*j - 2.2, j=0..11.
__device__ __forceinline__ void bspline8(float xv, float* out) {
  float b[11];
  #pragma unroll
  for (int j = 0; j < 11; ++j) {
    float g0 = 0.4f * (float)j - 2.2f;
    b[j] = (xv >= g0 && xv < g0 + 0.4f) ? 1.0f : 0.0f;
  }
  #pragma unroll
  for (int k = 1; k <= 3; ++k) {
    float inv = 1.0f / (0.4f * (float)k);
    #pragma unroll
    for (int j = 0; j + k < 11; ++j) {
      float gj = 0.4f * (float)j - 2.2f;
      float gk = 0.4f * (float)(j + k + 1) - 2.2f;
      b[j] = (xv - gj) * inv * b[j] + (gk - xv) * inv * b[j + 1];
    }
  }
  #pragma unroll
  for (int c = 0; c < 8; ++c) out[c] = b[c];
}

// ---------------------------------------------------------------------------
// TDM: DMA a 2D f16 tile [rows x cols] (memory row stride = row_stride elts)
// into LDS at lds_off, with hardware row padding:
//   pad_interval code: pad after 2^(code+1) DWORDs; pad_amount code: +code+1 DW.
// D# per CDNA5 ISA; tracked by TENSORcnt.
// ---------------------------------------------------------------------------
__device__ __forceinline__ void tdm_load_tile_f16(
    const _Float16* gbase, unsigned lds_off,
    unsigned cols, unsigned rows, unsigned row_stride,
    unsigned pad_interval_code, unsigned pad_amount_code) {
  unsigned long long ga = (unsigned long long)(uintptr_t)gbase;
  u32x4 g0;
  g0[0] = 1u;                                   // count=1, user descriptor
  g0[1] = lds_off;                              // lds_addr (bytes)
  g0[2] = (unsigned)ga;                         // global_addr[31:0]
  g0[3] = (unsigned)((ga >> 32) & 0x1FFFFFFu)   // global_addr[56:32]
          | (2u << 30);                         // type = 2 ("image")
  i32x8 g1;
  g1[0] = (int)((1u << 16)                      // data_size = 1 -> 2 bytes
              | (1u << 20)                      // pad_enable
              | (pad_interval_code << 22)
              | (pad_amount_code << 25));
  g1[1] = (int)(row_stride << 16);              // tensor_dim0 (bits 79:48)
  g1[2] = (int)(rows << 16);                    // tensor_dim1 (bits 111:80)
  g1[3] = (int)(cols << 16);                    // tile_dim0   (bits 127:112)
  g1[4] = (int)rows;                            // tile_dim1   (bits 143:128)
  g1[5] = (int)row_stride;                      // tensor_dim0_stride
  g1[6] = 0;
  g1[7] = 0;
  i32x4 gz = {0, 0, 0, 0};                      // 2D tile: groups 2/3 unused
#if defined(__clang_major__) && (__clang_major__ >= 23)
  i32x8 gz8 = {0, 0, 0, 0, 0, 0, 0, 0};
  __builtin_amdgcn_tensor_load_to_lds(g0, g1, gz, gz, gz8, 0);
#else
  __builtin_amdgcn_tensor_load_to_lds(g0, g1, gz, gz, 0);
#endif
}

// ---------------------------------------------------------------------------
// Kernel 0: fold spline_scaler into spline_w, concat with base_w, cast f16.
// W1c: [K1][H_MID], W2c: [K2][N_OUT]  (K-major, N contiguous)
// ---------------------------------------------------------------------------
__global__ __launch_bounds__(256) void prep_weights(
    const float* __restrict__ bw1, const float* __restrict__ sw1, const float* __restrict__ ss1,
    const float* __restrict__ bw2, const float* __restrict__ sw2, const float* __restrict__ ss2,
    _Float16* __restrict__ W1c, _Float16* __restrict__ W2c) {
  int idx = blockIdx.x * 256 + threadIdx.x;
  const int n2 = K2 * N_OUT;
  if (idx < n2) {
    int k = idx / N_OUT, n = idx % N_OUT;
    float v;
    if (k < H_MID) {
      v = bw2[n * H_MID + k];
    } else {
      int kk = k - H_MID, i = kk >> 3, c = kk & 7;
      v = sw2[(n * H_MID + i) * 8 + c] * ss2[n * H_MID + i];
    }
    W2c[idx] = (_Float16)v;
  } else {
    int j = idx - n2;
    if (j < K1 * H_MID) {
      int k = j / H_MID, o = j % H_MID;
      float v;
      if (k < F_IN) {
        v = bw1[o * F_IN + k];
      } else {
        int kk = k - F_IN, i = kk >> 3, c = kk & 7;
        v = sw1[(o * F_IN + i) * 8 + c] * ss1[o * F_IN + i];
      }
      W1c[j] = (_Float16)v;
    }
  }
}

// ---------------------------------------------------------------------------
// Kernel 1: layernorm + KAN features + (32x288)x(288x64) WMMA GEMM -> h2 f32
// Block = 256 threads (8 waves), 32 rows per block.
// W1c tile DMA'd by TDM (overlaps the feature build).
// ---------------------------------------------------------------------------
#define A1_STRIDE 296   // 288 + 8 pad (f16): 16B-aligned rows, conflict-spread
#define W1_STRIDE 72    // 64 f16 + 16B TDM pad = 144B rows

__global__ __launch_bounds__(256) void layer1_kernel(
    const float* __restrict__ x, const float* __restrict__ ln_w, const float* __restrict__ ln_b,
    const _Float16* __restrict__ W1c, float* __restrict__ h2) {
  __shared__ _Float16 A1[32 * A1_STRIDE];
  __shared__ _Float16 W1s[K1 * W1_STRIDE];
  __shared__ float mu_s[32], rs_s[32];

  const int tid  = threadIdx.x;
  const int row0 = blockIdx.x * 32;
  const int w = tid >> 5, lane = tid & 31;

  // Kick off the W1c tile DMA: 288 rows x 64 f16 (128B) -> 144B LDS rows.
  // pad_interval code 4 = 32 DWORDs (128B); pad_amount code 3 = 4 DW (16B).
  if (w == 0)
    tdm_load_tile_f16(W1c, (unsigned)(uintptr_t)W1s,
                      /*cols=*/H_MID, /*rows=*/K1, /*row_stride=*/H_MID,
                      /*pad_interval=*/4u, /*pad_amount=*/3u);

  // Per-row layernorm statistics (one thread per row).
  if (tid < 32) {
    const float* xr = x + (size_t)(row0 + tid) * F_IN;
    float s = 0.f, ss = 0.f;
    #pragma unroll
    for (int i = 0; i < F_IN; ++i) { float v = xr[i]; s += v; ss += v * v; }
    float mu  = s * (1.0f / F_IN);
    float var = ss * (1.0f / F_IN) - mu * mu;
    mu_s[tid] = mu;
    rs_s[tid] = rsqrtf(var + 1e-5f);
  }
  __syncthreads();

  // Build f16 feature tile: [silu(h) (32) | bsplines(h) (32*8)] per row.
  for (int p = tid; p < 32 * F_IN; p += 256) {
    int row = p >> 5, i = p & 31;
    float xv = x[(size_t)(row0 + row) * F_IN + i];
    float h = (xv - mu_s[row]) * rs_s[row] * ln_w[i] + ln_b[i];
    A1[row * A1_STRIDE + i] = (_Float16)silu_f(h);
    float bs[8];
    bspline8(h, bs);
    #pragma unroll
    for (int c = 0; c < 8; ++c)
      A1[row * A1_STRIDE + F_IN + i * 8 + c] = (_Float16)bs[c];
  }
  if (w == 0) __builtin_amdgcn_s_wait_tensorcnt(0);  // W1s DMA complete
  __syncthreads();

  // 8 waves cover the 32x64 output as 2(M) x 4(N) tiles of 16x16.
  const int mt = w & 1, nt = w >> 1;
  const int hsel = lane >> 4;
  const int lrow = (mt * 16 + (lane & 15)) * A1_STRIDE;
  f32x8 acc0 = {}, acc1 = {};   // two chains for XDL ILP
  #pragma unroll
  for (int kc = 0; kc < K1 / 32; ++kc) {
    half8_t alo = *(const half8_t*)&A1[lrow + kc * 32 + hsel * 8];
    half8_t ahi = *(const half8_t*)&A1[lrow + kc * 32 + 16 + hsel * 8];
    const int nn = nt * 16 + (lane & 15);
    half8_t blo, bhi;
    #pragma unroll
    for (int s = 0; s < 8; ++s) {
      blo[s] = W1s[(kc * 32 + hsel * 8 + s) * W1_STRIDE + nn];
      bhi[s] = W1s[(kc * 32 + 16 + hsel * 8 + s) * W1_STRIDE + nn];
    }
    if (kc & 1)
      acc1 = __builtin_amdgcn_wmma_f32_16x16x32_f16(
          false, cat8(alo, ahi), false, cat8(blo, bhi), (short)0, acc1, false, false);
    else
      acc0 = __builtin_amdgcn_wmma_f32_16x16x32_f16(
          false, cat8(alo, ahi), false, cat8(blo, bhi), (short)0, acc0, false, false);
  }
  const int nb = nt * 16 + (lane & 15);
  const int mb = row0 + mt * 16 + (lane >> 4) * 8;
  #pragma unroll
  for (int r = 0; r < 8; ++r)
    h2[(size_t)(mb + r) * H_MID + nb] = acc0[r] + acc1[r];
}

// ---------------------------------------------------------------------------
// Kernel 2: KAN features of h2 + (64x576)x(576x4096) WMMA GEMM -> out f32
// Block = 256 threads (8 waves), 64 rows per block; A register-resident;
// W panels streamed by TDM into double-buffered LDS.
// ---------------------------------------------------------------------------
#define A2_STRIDE 584   // 576 + 8 pad (f16): 16B-aligned rows
#define W2_STRIDE 40    // 32 f16 + 16B TDM pad = 80B rows

__global__ __launch_bounds__(256) void layer2_kernel(
    const float* __restrict__ h2, const _Float16* __restrict__ W2c,
    float* __restrict__ out) {
  extern __shared__ _Float16 lds[];
  _Float16* A2  = lds;                      // [64][A2_STRIDE] (feature phase)
  _Float16* Wb0 = lds;                      // panel buffer 0 (aliases A2)
  _Float16* Wb1 = lds + K2 * W2_STRIDE;     // panel buffer 1

  const int tid  = threadIdx.x;
  const int row0 = blockIdx.x * 64;
  const int w = tid >> 5, lane = tid & 31;

  // Phase 1: build f16 feature tile [silu(64) | bsplines(64*8)] for 64 rows.
  for (int p = tid; p < 64 * H_MID; p += 256) {
    int row = p >> 6, i = p & 63;
    float v = h2[(size_t)(row0 + row) * H_MID + i];
    A2[row * A2_STRIDE + i] = (_Float16)silu_f(v);
    float bs[8];
    bspline8(v, bs);
    #pragma unroll
    for (int c = 0; c < 8; ++c)
      A2[row * A2_STRIDE + H_MID + i * 8 + c] = (_Float16)bs[c];
  }
  __syncthreads();

  // Phase 2a: each wave copies its 16-row A tile into registers (18 frags).
  const int mt = w >> 1, ns = w & 1;     // 4 M-tiles x 2 N-subtiles
  const int hsel = lane >> 4;
  const int lrow = (mt * 16 + (lane & 15)) * A2_STRIDE;
  half16_t areg[K2 / 32];
  #pragma unroll
  for (int kc = 0; kc < K2 / 32; ++kc) {
    half8_t lo = *(const half8_t*)&A2[lrow + kc * 32 + hsel * 8];
    half8_t hi = *(const half8_t*)&A2[lrow + kc * 32 + 16 + hsel * 8];
    areg[kc] = cat8(lo, hi);
  }
  __syncthreads();   // A in registers; LDS reusable for TDM panel buffers.

  // LDS byte offsets of the panel buffers (LDS aperture: offset = addr[31:0]).
  const unsigned wb_off0 = (unsigned)(uintptr_t)Wb0;
  const unsigned wb_off1 = (unsigned)(uintptr_t)Wb1;

  const int NIT = N_OUT / 32;   // 128 panels
  // Preload panel 0: [K2 x 32] f16, 64B rows padded to 80B.
  // pad_interval code 3 = 16 DWORDs (64B); pad_amount code 3 = 4 DW (16B).
  if (w == 0)
    tdm_load_tile_f16(W2c, wb_off0, 32u, (unsigned)K2, (unsigned)N_OUT, 3u, 3u);

  for (int n_iter = 0; n_iter < NIT; ++n_iter) {
    const int n0 = n_iter * 32;
    if (w == 0) {
      if (n_iter + 1 < NIT) {
        // issue next panel into the other buffer, then wait for current one
        tdm_load_tile_f16(W2c + (size_t)(n_iter + 1) * 32,
                          (n_iter & 1) ? wb_off0 : wb_off1,
                          32u, (unsigned)K2, (unsigned)N_OUT, 3u, 3u);
        __builtin_amdgcn_s_wait_tensorcnt(1);
      } else {
        __builtin_amdgcn_s_wait_tensorcnt(0);
      }
    }
    __syncthreads();   // panel n_iter visible to all waves

    const _Float16* Wb = (n_iter & 1) ? Wb1 : Wb0;
    f32x8 acc0 = {}, acc1 = {};   // two accumulation chains for XDL ILP
    #pragma unroll
    for (int kc = 0; kc < K2 / 32; ++kc) {
      const int nn = ns * 16 + (lane & 15);
      half8_t blo, bhi;
      #pragma unroll
      for (int s = 0; s < 8; ++s) {
        blo[s] = Wb[(kc * 32 + hsel * 8 + s) * W2_STRIDE + nn];
        bhi[s] = Wb[(kc * 32 + 16 + hsel * 8 + s) * W2_STRIDE + nn];
      }
      if (kc & 1)
        acc1 = __builtin_amdgcn_wmma_f32_16x16x32_f16(
            false, areg[kc], false, cat8(blo, bhi), (short)0, acc1, false, false);
      else
        acc0 = __builtin_amdgcn_wmma_f32_16x16x32_f16(
            false, areg[kc], false, cat8(blo, bhi), (short)0, acc0, false, false);
    }

    const int nb = n0 + ns * 16 + (lane & 15);
    const int mb = row0 + mt * 16 + (lane >> 4) * 8;
    #pragma unroll
    for (int r = 0; r < 8; ++r)
      out[(size_t)(mb + r) * N_OUT + nb] = acc0[r] + acc1[r];

    __syncthreads();   // all reads of buf[n&1] done before it is refilled
  }
}

// ---------------------------------------------------------------------------
// Host-side launcher.
// Workspace layout: [W1c f16 36.9KB][pad][W2c f16 4.5MB][pad][h2 f32 16MB]
// ---------------------------------------------------------------------------
extern "C" void kernel_launch(void* const* d_in, const int* in_sizes, int n_in,
                              void* d_out, int out_size, void* d_ws, size_t ws_size,
                              hipStream_t stream) {
  const float* x    = (const float*)d_in[0];
  const float* ln_w = (const float*)d_in[1];
  const float* ln_b = (const float*)d_in[2];
  const float* bw1  = (const float*)d_in[3];
  const float* sw1  = (const float*)d_in[4];
  const float* ss1  = (const float*)d_in[5];
  const float* bw2  = (const float*)d_in[6];
  const float* sw2  = (const float*)d_in[7];
  const float* ss2  = (const float*)d_in[8];
  // d_in[9] = grid1, d_in[10] = grid2: uniform grids recomputed as constants.
  float* out = (float*)d_out;

  char* ws = (char*)d_ws;
  _Float16* W1c = (_Float16*)ws;                                    // 36,864 B
  _Float16* W2c = (_Float16*)(ws + 64 * 1024);                      // 4,718,592 B
  float*    h2  = (float*)(ws + 64 * 1024 + 5 * 1024 * 1024);       // 16,777,216 B

  const int prep_n = K2 * N_OUT + K1 * H_MID;
  prep_weights<<<(prep_n + 255) / 256, 256, 0, stream>>>(
      bw1, sw1, ss1, bw2, sw2, ss2, W1c, W2c);

  layer1_kernel<<<B_ROWS / 32, 256, 0, stream>>>(x, ln_w, ln_b, W1c, h2);

  // Dynamic LDS: max(feature tile 64*584*2 = 74,752 B,
  //                  2 TDM panel buffers 2*576*40*2 = 92,160 B)
  const size_t lds2 = (size_t)2 * K2 * W2_STRIDE * sizeof(_Float16);
  layer2_kernel<<<B_ROWS / 64, 256, lds2, stream>>>(h2, W2c, out);
}
